// MBS_Net_Optimized_14147622273750
// MI455X (gfx1250) — compile-verified, matrix operand order checked by reading, not run on hardware
//
#include <hip/hip_runtime.h>
#include <hip/hip_bf16.h>
#include <math.h>

// ---------------- CDNA5 / gfx1250 types ----------------
typedef __attribute__((ext_vector_type(16))) _Float16 v16h;
typedef __attribute__((ext_vector_type(8)))  _Float16 v8h;
typedef __attribute__((ext_vector_type(8)))  float    v8f;

#define N_DIM   128
#define K_BANDS 30
#define T_LEN   640
#define B_BATCH 2
#define DSTATE  12
#define DTRANK  8
#define R_ROWS  (B_BATCH * K_BANDS * T_LEN)   // 38400 rows of the (b,k,t) flattening

// ---------------- f32 -> f16 weight pre-conversion ----------------
__global__ void f32_to_f16_kernel(const float* __restrict__ s,
                                  _Float16* __restrict__ d, int n) {
  const int i = blockIdx.x * blockDim.x + threadIdx.x;
  if (i < n) d[i] = (_Float16)s[i];
}

// =====================================================================
// WMMA GEMM, strip-mined:  C[batch] = act( A[batch] @ W[batch%wmod]^T + bias + resid )
// A: (M,Kd) f32 row-major.  W: (N,Kd) f16 row-major (pre-converted).
// One wave -> 16 x (TN*16) strip of C: TN f32 accumulators, A fragment
// converted once per k-step and reused by TN consecutive WMMAs.
// Requires M%16==0, Kd%32==0, N % (TN*16)==0.
// ACT: 0 none, 1 sigmoid, 2 tanh.
// =====================================================================
template <int TN, int ACT>
__global__ void wmma_gemm(const float* __restrict__ A, long long aBatch,
                          const _Float16* __restrict__ W, long long wBatch, int wmod,
                          const float* __restrict__ bias, long long biasBatch,
                          const float* __restrict__ resid,
                          float* __restrict__ C, long long cBatch,
                          int M, int N, int Kd) {
  const int wave = threadIdx.x >> 5;
  const int lane = threadIdx.x & 31;
  const int tilesM  = M >> 4;
  const int nStrips = N / (TN * 16);
  const int widx = blockIdx.x * 4 + wave;
  if (widx >= tilesM * nStrips) return;          // wave-uniform: EXEC all-ones
  const int mt    = widx / nStrips;
  const int nBase = (widx % nStrips) * (TN * 16);
  const int batch = blockIdx.y;

  const float*    Ab = A + (long long)batch * aBatch;
  const _Float16* Wb = W + (long long)(batch % wmod) * wBatch;
  const float*    Bb = bias ? bias + (long long)(batch % wmod) * biasBatch : nullptr;
  float*          Cb = C + (long long)batch * cBatch;
  const float*    Rb = resid ? resid + (long long)batch * cBatch : nullptr;

  // ISA 16-bit A/B fragment layout (16x32):
  //   lanes 0-15  : row = lane,     K in {0..7} U {16..23}
  //   lanes 16-31 : row = lane-16,  K in {8..15} U {24..31}
  const int lr = lane & 15;
  const int kb = (lane & 16) ? 8 : 0;
  const float* arow = Ab + (long long)(mt * 16 + lr) * Kd + kb;

  v8f acc[TN];
#pragma unroll
  for (int j = 0; j < TN; ++j) acc[j] = (v8f){0.f,0.f,0.f,0.f,0.f,0.f,0.f,0.f};

  for (int kk = 0; kk < Kd; kk += 32) {
    if (kk + 32 < Kd)                            // global_prefetch_b8 (next A slice)
      __builtin_prefetch(arow + kk + 32, 0, 1);

    // ---- A fragment: 4x float4 loads, 8x v_cvt_pk_f16_f32, reused TN times
    const float4 a0 = *(const float4*)(arow + kk);
    const float4 a1 = *(const float4*)(arow + kk + 4);
    const float4 a2 = *(const float4*)(arow + kk + 16);
    const float4 a3 = *(const float4*)(arow + kk + 20);
    v16h af;
    af[0]=(_Float16)a0.x; af[1]=(_Float16)a0.y; af[2]=(_Float16)a0.z; af[3]=(_Float16)a0.w;
    af[4]=(_Float16)a1.x; af[5]=(_Float16)a1.y; af[6]=(_Float16)a1.z; af[7]=(_Float16)a1.w;
    af[8]=(_Float16)a2.x; af[9]=(_Float16)a2.y; af[10]=(_Float16)a2.z; af[11]=(_Float16)a2.w;
    af[12]=(_Float16)a3.x; af[13]=(_Float16)a3.y; af[14]=(_Float16)a3.z; af[15]=(_Float16)a3.w;

#pragma unroll
    for (int j = 0; j < TN; ++j) {
      const _Float16* wrow = Wb + (long long)(nBase + j * 16 + lr) * Kd + kb + kk;
      const v8h lo = *(const v8h*)(wrow);        // 16B load, no conversion
      const v8h hi = *(const v8h*)(wrow + 16);   // 16B load
      v16h bf;
#pragma unroll
      for (int q = 0; q < 8; ++q) { bf[q] = lo[q]; bf[q + 8] = hi[q]; }
      acc[j] = __builtin_amdgcn_wmma_f32_16x16x32_f16(
          false, af, false, bf, (short)0, acc[j], false, false);
    }
  }

  // ---- epilogue.  C/D layout: VGPR i -> lanes 0-15: M=i ; lanes 16-31: M=i+8
  const int rbase = mt * 16 + ((lane & 16) ? 8 : 0);
#pragma unroll
  for (int j = 0; j < TN; ++j) {
    const int col = nBase + j * 16 + (lane & 15);
    const float bv = Bb ? Bb[col] : 0.f;
#pragma unroll
    for (int i = 0; i < 8; ++i) {
      long long off = (long long)(rbase + i) * N + col;
      float v = acc[j][i] + bv;
      if (Rb) v += Rb[off];
      if (ACT == 1) v = 1.f / (1.f + __expf(-v));
      else if (ACT == 2) v = tanhf(v);
      Cb[off] = v;
    }
  }
}

// =====================================================================
// Band-split feature build + LayerNorm over 32 features.
// Row r = (b*K + k)*T + t ; lane = feature f (0..31).
// =====================================================================
__global__ void feat_kernel(const float* __restrict__ x,
                            const float* __restrict__ lnw,
                            const float* __restrict__ lnb,
                            float* __restrict__ feat) {
  const int row  = blockIdx.x * 4 + (threadIdx.x >> 5);
  const int lane = threadIdx.x & 31;
  if (row >= R_ROWS) return;
  const int t  = row % T_LEN;
  const int bk = row / T_LEN;
  const int k  = bk % K_BANDS;
  const int b  = bk / K_BANDS;
  const int comp = lane >> 4;                    // 0 = real, 1 = imag
  const int ch   = k * 16 + (lane & 15);
  float v = x[(((long long)b * 2 + comp) * 480 + ch) * T_LEN + t];

  float s = v;
#pragma unroll
  for (int m = 16; m >= 1; m >>= 1) s += __shfl_xor(s, m, 32);
  const float mu = s * (1.f / 32.f);
  float d = v - mu, q = d * d;
#pragma unroll
  for (int m = 16; m >= 1; m >>= 1) q += __shfl_xor(q, m, 32);
  const float rstd = rsqrtf(q * (1.f / 32.f) + 1e-5f);
  feat[(long long)row * 32 + lane] = d * rstd * lnw[k * 32 + lane] + lnb[k * 32 + lane];
}

// =====================================================================
// LayerNorm over 128 (optional x+add first, optional exact GELU after).
// One wave per row, float4 per lane.
// =====================================================================
__global__ void ln128_kernel(const float* __restrict__ x,
                             const float* __restrict__ addend,
                             const float* __restrict__ w,
                             const float* __restrict__ b,
                             float* __restrict__ out, int R, int gelu) {
  const int r    = blockIdx.x * 4 + (threadIdx.x >> 5);
  const int lane = threadIdx.x & 31;
  if (r >= R) return;
  float4 v = ((const float4*)(x + (long long)r * N_DIM))[lane];
  if (addend) {
    float4 a = ((const float4*)(addend + (long long)r * N_DIM))[lane];
    v.x += a.x; v.y += a.y; v.z += a.z; v.w += a.w;
  }
  float s = v.x + v.y + v.z + v.w;
#pragma unroll
  for (int m = 16; m >= 1; m >>= 1) s += __shfl_xor(s, m, 32);
  const float mu = s * (1.f / 128.f);
  float dx = v.x - mu, dy = v.y - mu, dz = v.z - mu, dw = v.w - mu;
  float q = dx * dx + dy * dy + dz * dz + dw * dw;
#pragma unroll
  for (int m = 16; m >= 1; m >>= 1) q += __shfl_xor(q, m, 32);
  const float rstd = rsqrtf(q * (1.f / 128.f) + 1e-5f);
  const int c = lane * 4;
  float o[4] = {dx, dy, dz, dw};
#pragma unroll
  for (int j = 0; j < 4; ++j) {
    float y = o[j] * rstd * w[c + j] + b[c + j];
    if (gelu) y = 0.5f * y * (1.f + erff(y * 0.70710678118f));
    o[j] = y;
  }
  float4 ov = {o[0], o[1], o[2], o[3]};
  ((float4*)(out + (long long)r * N_DIM))[lane] = ov;
}

// ---------------- depthwise causal conv (DCONV=4) + bias + SiLU ----------------
__global__ void conv_silu_kernel(const float* __restrict__ xz,
                                 const float* __restrict__ cw,
                                 const float* __restrict__ cb,
                                 float* __restrict__ xp, int total) {
  const int idx = blockIdx.x * blockDim.x + threadIdx.x;
  if (idx >= total) return;
  const int d = idx & (N_DIM - 1);
  const int r = idx >> 7;
  const int t = r % T_LEN;
  float acc = cb[d];
#pragma unroll
  for (int j = 0; j < 4; ++j) {
    const int tt = t - 3 + j;
    if (tt >= 0) acc += xz[(long long)(r - 3 + j) * 256 + d] * cw[d * 4 + j];
  }
  acc = acc / (1.f + __expf(-acc));              // SiLU
  xp[idx] = acc;
}

// ---------------- dt = softplus(dtr @ dt_w^T + dt_b) ----------------
__global__ void dt_kernel(const float* __restrict__ dbl,
                          const float* __restrict__ dtw,
                          const float* __restrict__ dtb,
                          float* __restrict__ out, int total) {
  const int idx = blockIdx.x * blockDim.x + threadIdx.x;
  if (idx >= total) return;
  const int d = idx & (N_DIM - 1);
  const int r = idx >> 7;
  float acc = dtb[d];
#pragma unroll
  for (int j = 0; j < DTRANK; ++j)
    acc += dbl[(long long)r * 32 + j] * dtw[d * DTRANK + j];
  out[idx] = (acc > 20.f) ? acc : log1pf(__expf(acc));
}

// =====================================================================
// Selective scan: one block per sequence (60), one thread per channel (128),
// 12-element state in registers. Fused y = (scan + u*D) * silu(zg).
// =====================================================================
__global__ void scan_kernel(const float* __restrict__ xp,
                            const float* __restrict__ dtv,
                            const float* __restrict__ dbl,
                            const float* __restrict__ xz,
                            const float* __restrict__ A_log,
                            const float* __restrict__ Dp,
                            float* __restrict__ y) {
  const int s = blockIdx.x;
  const int d = threadIdx.x;
  float Areg[DSTATE], h[DSTATE];
#pragma unroll
  for (int n = 0; n < DSTATE; ++n) {
    Areg[n] = -__expf(A_log[d * DSTATE + n]);
    h[n] = 0.f;
  }
  const float Dd = Dp[d];
  for (int t = 0; t < T_LEN; ++t) {
    const long long r   = (long long)s * T_LEN + t;
    const long long idx = r * N_DIM + d;
    const float u  = xp[idx];
    const float dt = dtv[idx];
    const float* bc = dbl + r * 32;              // cols 8..19 = B, 20..31 = C
    float acc = 0.f;
#pragma unroll
    for (int n = 0; n < DSTATE; ++n) {
      const float dA = __expf(dt * Areg[n]);
      h[n] = dA * h[n] + dt * bc[8 + n] * u;
      acc += h[n] * bc[20 + n];
    }
    float yv = acc + u * Dd;
    const float z = xz[r * 256 + 128 + d];
    yv *= z / (1.f + __expf(-z));                // * silu(zg)
    y[idx] = yv;
  }
}

// ---------------- elementwise multiply (zm = z * mag) ----------------
__global__ void mul_kernel(const float* __restrict__ a,
                           const float* __restrict__ b,
                           float* __restrict__ o, int total) {
  const int idx = blockIdx.x * blockDim.x + threadIdx.x;
  if (idx >= total) return;
  o[idx] = a[idx] * b[idx];
}

// ---------------- enhanced[b,m,t] = sum_k tmp[(b*K+k)*T+t, m] + sum_k bm_b[k,m] --
__global__ void bm_reduce_kernel(const float* __restrict__ tmp,
                                 const float* __restrict__ bm_b,
                                 float* __restrict__ out) {
  const int idx = blockIdx.x * blockDim.x + threadIdx.x;
  if (idx >= B_BATCH * N_DIM * T_LEN) return;
  const int t = idx % T_LEN;
  const int m = (idx / T_LEN) % N_DIM;
  const int b = idx / (T_LEN * N_DIM);
  float acc = 0.f;
  for (int k = 0; k < K_BANDS; ++k) {
    acc += tmp[((long long)(b * K_BANDS + k) * T_LEN + t) * N_DIM + m];
    acc += bm_b[k * N_DIM + m];
  }
  out[((long long)b * N_DIM + m) * T_LEN + t] = acc;
}

// =====================================================================
// Host-side orchestration
// =====================================================================
extern "C" void kernel_launch(void* const* d_in, const int* in_sizes, int n_in,
                              void* d_out, int out_size, void* d_ws, size_t ws_size,
                              hipStream_t stream) {
  const float* X        = (const float*)d_in[0];
  const float* bs_ln_w  = (const float*)d_in[1];
  const float* bs_ln_b  = (const float*)d_in[2];
  const float* bs_fc_w  = (const float*)d_in[3];
  const float* bs_fc_b  = (const float*)d_in[4];
  const float* m_ln_w   = (const float*)d_in[5];
  const float* m_ln_b   = (const float*)d_in[6];
  const float* m_in_w   = (const float*)d_in[7];
  const float* m_conv_w = (const float*)d_in[8];
  const float* m_conv_b = (const float*)d_in[9];
  const float* m_xproj_w= (const float*)d_in[10];
  const float* m_dt_w   = (const float*)d_in[11];
  const float* m_dt_b   = (const float*)d_in[12];
  const float* m_A_log  = (const float*)d_in[13];
  const float* m_D      = (const float*)d_in[14];
  const float* m_out_w  = (const float*)d_in[15];
  const float* cb_w     = (const float*)d_in[16];
  const float* cb_b     = (const float*)d_in[17];
  const float* cb_ln_w  = (const float*)d_in[18];
  const float* cb_ln_b  = (const float*)d_in[19];
  const float* cbn_w    = (const float*)d_in[20];
  const float* cbn_b    = (const float*)d_in[21];
  const float* mag_w    = (const float*)d_in[22];
  const float* mag_b    = (const float*)d_in[23];
  const float* dec_w1   = (const float*)d_in[24];
  const float* dec_b1   = (const float*)d_in[25];
  const float* dec_w2   = (const float*)d_in[26];
  const float* dec_b2   = (const float*)d_in[27];
  const float* bm_w     = (const float*)d_in[28];
  const float* bm_b     = (const float*)d_in[29];
  float* out = (float*)d_out;

  // ---- f32 workspace (~162 MB, L2-resident on MI455X)
  float* ws = (float*)d_ws;
  const long long R = R_ROWS;
  long long o = 0;
  float* zb   = ws + o; o += R * 128;   // z, kept for zm
  float* xbb  = ws + o; o += R * 128;   // residual stream xb
  float* xnb  = ws + o; o += R * 128;   // LN out / t1 / mag
  float* xzb  = ws + o; o += R * 256;   // in_proj out (xp_pre | zg) / dec1 out
  float* xpb  = ws + o; o += R * 128;   // conv out / feats / dec2 out
  float* dtb  = ws + o; o += R * 128;   // dt / oc2
  float* ybuf = ws + o; o += R * 128;   // scan out / zm / bm tmp
  float* dblb = ws + o; o += R * 32;    // feat (32) / xproj out (32)

  // ---- f16 weight area (~1.8 MB)
  _Float16* hw = (_Float16*)(ws + o);
  long long h = 0;
  _Float16* h_bsfc = hw + h; h += (long long)K_BANDS * 128 * 32;
  _Float16* h_in   = hw + h; h += 4LL * 256 * 128;
  _Float16* h_xp   = hw + h; h += 4LL * 32 * 128;
  _Float16* h_out  = hw + h; h += 4LL * 128 * 128;
  _Float16* h_cb   = hw + h; h += 128LL * 128;
  _Float16* h_mag  = hw + h; h += 128LL * 128;
  _Float16* h_d1   = hw + h; h += 256LL * 128;
  _Float16* h_d2   = hw + h; h += 128LL * 256;
  _Float16* h_bm   = hw + h; h += (long long)K_BANDS * 128 * 128;

  const int totalRN = (int)(R * 128);
  const dim3 blk128(128), blk256(256);
  auto cvt = [&](const float* s, _Float16* d, int n) {
    f32_to_f16_kernel<<<dim3((n + 255) / 256), blk256, 0, stream>>>(s, d, n);
  };

  // ---- one-time (per launch) weight conversions
  cvt(bs_fc_w,  h_bsfc, K_BANDS * 128 * 32);
  cvt(m_in_w,   h_in,   4 * 256 * 128);
  cvt(m_xproj_w,h_xp,   4 * 32 * 128);
  cvt(m_out_w,  h_out,  4 * 128 * 128);
  cvt(cb_w,     h_cb,   128 * 128);
  cvt(mag_w,    h_mag,  128 * 128);
  cvt(dec_w1,   h_d1,   256 * 128);
  cvt(dec_w2,   h_d2,   128 * 256);
  cvt(bm_w,     h_bm,   K_BANDS * 128 * 128);

  // 0) band-split features + LN(32)
  feat_kernel<<<dim3(R_ROWS / 4), blk128, 0, stream>>>(X, bs_ln_w, bs_ln_b, dblb);

  // 1) per-band fc: z = feat @ bs_fc_w[k]^T + bs_fc_b[k]   (M=640,N=128,Kd=32)
  wmma_gemm<8, 0><<<dim3(10, B_BATCH * K_BANDS), blk128, 0, stream>>>(
      dblb, (long long)T_LEN * 32, h_bsfc, 128LL * 32, K_BANDS,
      bs_fc_b, 128, nullptr, zb, (long long)T_LEN * 128, T_LEN, 128, 32);

  hipMemcpyAsync(xbb, zb, (size_t)R * 128 * sizeof(float),
                 hipMemcpyDeviceToDevice, stream);          // xb = z

  // 2) Mamba layers
  for (int i = 0; i < 4; ++i) {
    ln128_kernel<<<dim3(R_ROWS / 4), blk128, 0, stream>>>(
        xbb, nullptr, m_ln_w + i * 128, m_ln_b + i * 128, xnb, R_ROWS, 0);

    // in_proj: (R,128)->(R,256) ; 2400 M-tiles x 2 strips
    wmma_gemm<8, 0><<<dim3(1200, 1), blk128, 0, stream>>>(
        xnb, 0, h_in + (long long)i * 256 * 128, 0, 1, nullptr, 0, nullptr,
        xzb, 0, R_ROWS, 256, 128);

    conv_silu_kernel<<<dim3(totalRN / 256), blk256, 0, stream>>>(
        xzb, m_conv_w + i * 128 * 4, m_conv_b + i * 128, xpb, totalRN);

    // xproj: (R,128)->(R,32) ; TN=2 (strip == 32)
    wmma_gemm<2, 0><<<dim3(600, 1), blk128, 0, stream>>>(
        xpb, 0, h_xp + (long long)i * 32 * 128, 0, 1, nullptr, 0, nullptr,
        dblb, 0, R_ROWS, 32, 128);

    dt_kernel<<<dim3(totalRN / 256), blk256, 0, stream>>>(
        dblb, m_dt_w + i * 128 * 8, m_dt_b + i * 128, dtb, totalRN);

    scan_kernel<<<dim3(B_BATCH * K_BANDS), blk128, 0, stream>>>(
        xpb, dtb, dblb, xzb, m_A_log + i * 128 * DSTATE, m_D + i * 128, ybuf);

    // out_proj with fused residual: xb = xb + y @ out_w^T
    wmma_gemm<8, 0><<<dim3(600, 1), blk128, 0, stream>>>(
        ybuf, 0, h_out + (long long)i * 128 * 128, 0, 1, nullptr, 0, xbb,
        xbb, 0, R_ROWS, 128, 128);
  }

  // 3) post-processing head
  wmma_gemm<8, 0><<<dim3(600, 1), blk128, 0, stream>>>(          // t1 = oc@cb^T+b
      xbb, 0, h_cb, 0, 1, cb_b, 0, nullptr, xnb, 0, R_ROWS, 128, 128);
  ln128_kernel<<<dim3(R_ROWS / 4), blk128, 0, stream>>>(         // oc2 = gelu(LN(t1))
      xnb, nullptr, cb_ln_w, cb_ln_b, dtb, R_ROWS, 1);
  ln128_kernel<<<dim3(R_ROWS / 4), blk128, 0, stream>>>(         // feats = LN(oc+oc2)
      xbb, dtb, cbn_w, cbn_b, xpb, R_ROWS, 0);
  wmma_gemm<8, 1><<<dim3(600, 1), blk128, 0, stream>>>(          // mag = sigmoid(...)
      xpb, 0, h_mag, 0, 1, mag_b, 0, nullptr, xnb, 0, R_ROWS, 128, 128);
  mul_kernel<<<dim3(totalRN / 256), blk256, 0, stream>>>(        // zm = z*mag
      zb, xnb, ybuf, totalRN);
  wmma_gemm<8, 2><<<dim3(1200, 1), blk128, 0, stream>>>(         // d1 = tanh(...)
      ybuf, 0, h_d1, 0, 1, dec_b1, 0, nullptr, xzb, 0, R_ROWS, 256, 128);
  wmma_gemm<8, 0><<<dim3(600, 1), blk128, 0, stream>>>(          // h = d1@dec2^T+b
      xzb, 0, h_d2, 0, 1, dec_b2, 0, nullptr, xpb, 0, R_ROWS, 128, 256);
  wmma_gemm<8, 0><<<dim3(10, B_BATCH * K_BANDS), blk128, 0, stream>>>( // per-band bm
      xpb, (long long)T_LEN * 128, h_bm, 128LL * 128, K_BANDS,
      nullptr, 0, nullptr, ybuf, (long long)T_LEN * 128, T_LEN, 128, 128);
  bm_reduce_kernel<<<dim3((B_BATCH * N_DIM * T_LEN) / 256), blk256, 0, stream>>>(
      ybuf, bm_b, out);
}
